// CentroidsTripletLoss_16913581211856
// MI455X (gfx1250) — matrix operand order repaired and checked by reading, not compiled, not run on hardware
//
#include <hip/hip_runtime.h>

#define B_ROWS      4096
#define D_DIM       512
#define NUM_CLASSES 1000
#define CTILES      63          // ceil(1000/16)
#define ALPHA_F     0.1f
#define BETA_F      1.0f

// workspace layout (float offsets)
#define WS_SM   0               // 12288 per-row softmax losses
#define WS_TR   12288           // 4096 per-row triplet losses
#define WS_CE   16384           // 4096 per-row center losses
#define WS_ESQ  20480           // 1000 exemplar squared norms (+pad)
#define WS_ASQ  21488           // 4096 anchor squared norms

typedef __attribute__((ext_vector_type(16))) __bf16 v16bf;
typedef __attribute__((ext_vector_type(8)))  __bf16 v8bf;
typedef __attribute__((ext_vector_type(8)))  float  v8f;
typedef __attribute__((ext_vector_type(4)))  float  v4f;

#define APITCH 520              // 512 + 8 bf16 pad -> kills LDS bank conflicts

__device__ inline float waveRedSum(float v) {
#pragma unroll
  for (int m = 16; m >= 1; m >>= 1) v += __shfl_xor(v, m, 32);
  return v;
}
__device__ inline float waveRedMax(float v) {
#pragma unroll
  for (int m = 16; m >= 1; m >>= 1) v = fmaxf(v, __shfl_xor(v, m, 32));
  return v;
}

// ---------------- squared norms of exemplar rows (0..999) and anchor rows (1000..5095)
__global__ void row_sqnorm_kernel(const float* __restrict__ exemplars,
                                  const float* __restrict__ anchor,
                                  float* __restrict__ ws) {
  const int lane = threadIdx.x & 31;
  const int row  = blockIdx.x * 8 + (threadIdx.x >> 5);
  if (row >= NUM_CLASSES + B_ROWS) return;
  const float* src;
  float* dst;
  if (row < NUM_CLASSES) { src = exemplars + (size_t)row * D_DIM;                 dst = ws + WS_ESQ + row; }
  else                   { src = anchor    + (size_t)(row - NUM_CLASSES) * D_DIM; dst = ws + WS_ASQ + (row - NUM_CLASSES); }
  float s = 0.f;
  for (int i = lane; i < D_DIM; i += 32) { float x = src[i]; s += x * x; }
  s = waveRedSum(s);
  if (lane == 0) *dst = s;
}

// ---------------- cross-entropy per row of outputs [3B, 1000]
__global__ void ce_rows_kernel(const float* __restrict__ outputs,
                               const int* __restrict__ la,
                               const int* __restrict__ ln,
                               float* __restrict__ ws_sm) {
  const int lane = threadIdx.x & 31;
  const int row  = blockIdx.x * 8 + (threadIdx.x >> 5);
  if (row >= 3 * B_ROWS) return;
  const float* x = outputs + (size_t)row * NUM_CLASSES;
  float m = -__builtin_inff();
  for (int i = lane; i < NUM_CLASSES; i += 32) m = fmaxf(m, x[i]);
  m = waveRedMax(m);
  float s = 0.f;
  for (int i = lane; i < NUM_CLASSES; i += 32) s += expf(x[i] - m);
  s = waveRedSum(s);
  if (lane == 0) {
    int label = (row < B_ROWS) ? la[row]
              : (row < 2 * B_ROWS) ? la[row - B_ROWS] : ln[row - 2 * B_ROWS];
    ws_sm[row] = -(x[label] - m - logf(s));
  }
}

// ---------------- triplet per row
__global__ void triplet_rows_kernel(const float* __restrict__ anchor,
                                    const float* __restrict__ positive,
                                    const float* __restrict__ negative,
                                    float* __restrict__ ws_tr) {
  const int lane = threadIdx.x & 31;
  const int row  = blockIdx.x * 8 + (threadIdx.x >> 5);
  if (row >= B_ROWS) return;
  const size_t base = (size_t)row * D_DIM;
  float sp = 0.f, sn = 0.f;
  for (int i = lane; i < D_DIM; i += 32) {
    float a = anchor[base + i];
    float dp = a - positive[base + i];
    float dn = a - negative[base + i];
    sp += dp * dp;
    sn += dn * dn;
  }
  sp = waveRedSum(sp);
  sn = waveRedSum(sn);
  if (lane == 0) ws_tr[row] = fmaxf(sqrtf(sp) - sqrtf(sn), 0.f);
}

// ---- async global->LDS copy of one exemplar tile (16 rows x 512 fp32 = 32KB)
// GVS addressing: mem = saddr(64) + vaddr(32) + ioffset; VDST = LDS byte address.
// Tracked by ASYNCcnt; masked lanes (padded classes) simply don't transfer.
__device__ inline void async_load_tile(int tid, int jt, float* stage,
                                       const float* __restrict__ exemplars) {
  unsigned long long sbase = (unsigned long long)exemplars;
  unsigned lds0 = (unsigned)(uintptr_t)stage;   // LDS aperture keeps offset in addr[31:0]
#pragma unroll
  for (int it = 0; it < 16; ++it) {
    int e   = it * 128 + tid;      // 16-byte chunk index within the 32KB tile
    int n   = e >> 7;              // exemplar row within tile
    int c16 = e & 127;             // 16B chunk within row
    int col = jt * 16 + n;
    unsigned lds  = lds0 + (unsigned)e * 16u;
    unsigned goff = (unsigned)col * 2048u + (unsigned)c16 * 16u;
    if (col < NUM_CLASSES) {
      asm volatile("global_load_async_to_lds_b128 %0, %1, %2"
                   :: "v"(lds), "v"(goff), "s"(sbase)
                   : "memory");
    }
  }
}

__device__ inline void wait_async0() {
  asm volatile("s_wait_asynccnt 0x0" ::: "memory");
}

__device__ inline v8f wmma_bf16(v16bf a, v16bf b, v8f c) {
  return __builtin_amdgcn_wmma_f32_16x16x32_bf16(false, a, false, b, (short)0, c, false, false);
}

// ---------------- center loss: anchor @ exemplars^T with split-bf16 WMMA
// block: 128 threads (4 waves); block tile = 64 anchor rows x all classes.
// B tiles are double-buffered through an fp32 LDS staging area filled by
// GLOBAL_LOAD_ASYNC_TO_LDS_B128 issued one tile ahead of the WMMA work.
__global__ void center_wmma_kernel(const float* __restrict__ anchor,
                                   const float* __restrict__ exemplars,
                                   const int* __restrict__ la,
                                   float* __restrict__ ws) {
  extern __shared__ char smem[];
  __bf16* sAhi   = (__bf16*)smem;                       // 64 * APITCH
  __bf16* sAlo   = sAhi + 64 * APITCH;
  __bf16* sBhi   = sAlo + 64 * APITCH;                  // 16 * APITCH
  __bf16* sBlo   = sBhi + 16 * APITCH;
  float*  stageF = (float*)(sBlo + 16 * APITCH);        // 2 * 16*512 fp32

  const int tid     = threadIdx.x;
  const int lane    = tid & 31;
  const int wave    = tid >> 5;
  const int halfsel = lane >> 4;              // 0 | 1
  const int nloc    = lane & 15;
  const int rowBase = blockIdx.x * 64;

  // kick off async copy of exemplar tile 0 -> overlaps with A staging below
  async_load_tile(tid, 0, stageF, exemplars);

  // ---- stage 64 anchor rows into LDS as bf16 hi + residual lo
  for (int idx = tid * 4; idx < 64 * D_DIM; idx += 128 * 4) {
    v4f x = *(const v4f*)(anchor + (size_t)rowBase * D_DIM + idx);
    int r = idx >> 9, k = idx & 511;
#pragma unroll
    for (int e = 0; e < 4; ++e) {
      __bf16 h = (__bf16)x[e];
      sAhi[r * APITCH + k + e] = h;
      sAlo[r * APITCH + k + e] = (__bf16)(x[e] - (float)h);
    }
  }

  // per-lane row metadata: accumulator row M = v + 8*halfsel within the wave's strip
  float asq_l[8];
  int   labv[8];
#pragma unroll
  for (int v = 0; v < 8; ++v) {
    int m = wave * 16 + v + 8 * halfsel;
    asq_l[v] = ws[WS_ASQ + rowBase + m];
    labv[v]  = la[rowBase + m];
  }

  float dmin[8], dref[8];
#pragma unroll
  for (int v = 0; v < 8; ++v) { dmin[v] = __builtin_inff(); dref[v] = 0.f; }

  const int m_op = wave * 16 + nloc;          // A operand row (lanes 0-15 == 16-31)

  for (int j = 0; j < CTILES; ++j) {
    wait_async0();                 // tile j fp32 bytes landed in stage[j&1]
    __syncthreads();               // + every wave done computing tile j-1

    // convert stage[j&1] fp32 -> bf16 hi/lo operand layout (zero-fill padded cols)
    {
      const float* st = stageF + (j & 1) * (16 * D_DIM);
      for (int idx = tid * 4; idx < 16 * D_DIM; idx += 128 * 4) {
        int n = idx >> 9, k = idx & 511;
        int col = j * 16 + n;
        v4f x = {0.f, 0.f, 0.f, 0.f};
        if (col < NUM_CLASSES) x = *(const v4f*)(st + idx);
#pragma unroll
        for (int e = 0; e < 4; ++e) {
          __bf16 h = (__bf16)x[e];
          sBhi[n * APITCH + k + e] = h;
          sBlo[n * APITCH + k + e] = (__bf16)(x[e] - (float)h);
        }
      }
    }
    __syncthreads();

    // prefetch tile j+1 asynchronously while the WMMAs below run
    if (j + 1 < CTILES)
      async_load_tile(tid, j + 1, stageF + ((j + 1) & 1) * (16 * D_DIM), exemplars);

    // three independent accumulator chains: hi*hi, hi*lo, lo*hi
    v8f chh = {0.f,0.f,0.f,0.f,0.f,0.f,0.f,0.f};
    v8f chl = {0.f,0.f,0.f,0.f,0.f,0.f,0.f,0.f};
    v8f clh = {0.f,0.f,0.f,0.f,0.f,0.f,0.f,0.f};
    for (int kb = 0; kb < D_DIM; kb += 32) {
      union U { v16bf v; v8bf h[2]; };
      U ah, al, bh, bl;
      // A layout (16-bit 16x32): lane-half K offset 8; chunks at k+0..7 and k+16..23
      const __bf16* pa  = sAhi + m_op * APITCH + kb + halfsel * 8;
      const __bf16* pal = sAlo + m_op * APITCH + kb + halfsel * 8;
      ah.h[0] = *(const v8bf*)(pa);       ah.h[1] = *(const v8bf*)(pa + 16);
      al.h[0] = *(const v8bf*)(pal);      al.h[1] = *(const v8bf*)(pal + 16);
      // B layout (16-bit 32x16): lane-half K offset 8; chunks at k+0..7 and k+16..23 (matches A)
      const __bf16* pb  = sBhi + nloc * APITCH + kb + halfsel * 8;
      const __bf16* pbl = sBlo + nloc * APITCH + kb + halfsel * 8;
      bh.h[0] = *(const v8bf*)(pb);       bh.h[1] = *(const v8bf*)(pb + 16);
      bl.h[0] = *(const v8bf*)(pbl);      bl.h[1] = *(const v8bf*)(pbl + 16);

      chh = wmma_bf16(ah.v, bh.v, chh);
      chl = wmma_bf16(ah.v, bl.v, chl);
      clh = wmma_bf16(al.v, bh.v, clh);
    }

    // epilogue: d2 = |a|^2 + |e|^2 - 2 a.e ; track min + labeled column
    int col  = j * 16 + nloc;
    float es = (col < NUM_CLASSES) ? ws[WS_ESQ + col] : __builtin_inff();
#pragma unroll
    for (int v = 0; v < 8; ++v) {
      float dot = chh[v] + (chl[v] + clh[v]);
      float d2  = asq_l[v] + es - 2.f * dot;
      dmin[v] = fminf(dmin[v], d2);
      if (labv[v] == col) dref[v] = d2;
    }
  }

  // reduce across the 16 lanes of each half-wave (rows differ between halves)
#pragma unroll
  for (int v = 0; v < 8; ++v) {
#pragma unroll
    for (int m = 8; m >= 1; m >>= 1) {
      dmin[v] = fminf(dmin[v], __shfl_xor(dmin[v], m, 32));
      dref[v] = fmaxf(dref[v], __shfl_xor(dref[v], m, 32));
    }
  }
  if (nloc == 0) {
#pragma unroll
    for (int v = 0; v < 8; ++v) {
      int m = wave * 16 + v + 8 * halfsel;
      float dr = sqrtf(fmaxf(dref[v], 0.f));
      float dc = sqrtf(fmaxf(dmin[v], 0.f));
      ws[WS_CE + rowBase + m] = fmaxf(dr - dc, 0.f);
    }
  }
}

// ---------------- deterministic final reduction
__global__ void finalize_kernel(const float* __restrict__ ws, float* __restrict__ out) {
  __shared__ float red[256];
  const int tid = threadIdx.x;
  float s_sm = 0.f, s_tr = 0.f, s_ce = 0.f;
  for (int i = tid; i < 3 * B_ROWS; i += 256) s_sm += ws[WS_SM + i];
  for (int i = tid; i < B_ROWS; i += 256) { s_tr += ws[WS_TR + i]; s_ce += ws[WS_CE + i]; }

  float totals[3];
  float vals[3] = { s_sm, s_tr, s_ce };
  for (int t = 0; t < 3; ++t) {
    red[tid] = vals[t];
    __syncthreads();
    for (int off = 128; off > 0; off >>= 1) {
      if (tid < off) red[tid] += red[tid + off];
      __syncthreads();
    }
    totals[t] = red[0];
    __syncthreads();
  }
  if (tid == 0) {
    float loss_softmax = totals[0] / (float)(3 * B_ROWS);
    float loss_triplet = totals[1];
    float loss_center  = totals[2];
    out[0] = loss_softmax + ALPHA_F * loss_center + BETA_F * loss_triplet;
    out[1] = loss_triplet;
    out[2] = loss_softmax;
    out[3] = loss_center;
  }
}

extern "C" void kernel_launch(void* const* d_in, const int* in_sizes, int n_in,
                              void* d_out, int out_size, void* d_ws, size_t ws_size,
                              hipStream_t stream) {
  const float* anchor    = (const float*)d_in[0];
  const float* positive  = (const float*)d_in[1];
  const float* negative  = (const float*)d_in[2];
  const float* outputs   = (const float*)d_in[3];
  const int*   la        = (const int*)d_in[4];
  const int*   ln        = (const int*)d_in[5];
  const float* exemplars = (const float*)d_in[6];
  float* out = (float*)d_out;
  float* ws  = (float*)d_ws;

  (void)in_sizes; (void)n_in; (void)out_size; (void)ws_size;

  // squared norms (exemplars + anchors): 5096 rows, wave per row
  row_sqnorm_kernel<<<(NUM_CLASSES + B_ROWS) / 8, 256, 0, stream>>>(exemplars, anchor, ws);
  // cross-entropy: 12288 rows, wave per row
  ce_rows_kernel<<<(3 * B_ROWS) / 8, 256, 0, stream>>>(outputs, la, ln, ws + WS_SM);
  // triplet: 4096 rows, wave per row
  triplet_rows_kernel<<<B_ROWS / 8, 256, 0, stream>>>(anchor, positive, negative, ws + WS_TR);
  // center: 64 blocks x 128 threads
  // dynamic LDS: A hi/lo (2*64*520 bf16) + B hi/lo (2*16*520 bf16) + fp32 stage (2*16*512 f32)
  const size_t smem_bytes = (size_t)(2 * 64 * APITCH + 2 * 16 * APITCH) * 2
                          + (size_t)(2 * 16 * D_DIM) * 4;   // 231,936 B
  center_wmma_kernel<<<B_ROWS / 64, 128, smem_bytes, stream>>>(anchor, exemplars, la, ws);
  // combine
  finalize_kernel<<<1, 256, 0, stream>>>(ws, out);
}
